// MHAModule_12489764896858
// MI455X (gfx1250) — compile-verified, matrix operand order checked by reading, not run on hardware
//
#include <hip/hip_runtime.h>
#include <hip/hip_bf16.h>

// Conformer-style relative-position MHA for MI455X (gfx1250, wave32, WMMA).
// - All O(N^3) math through v_wmma_f32_16x16x32_f16, f32 accumulate.
// - Big square GEMMs (QKV, out-proj) stage the weight panel in LDS via the
//   CDNA5 async path: global_load_async_to_lds_b128 + s_wait_asynccnt,
//   double-buffered, so WMMA B operands come from DS instead of VMEM.
// - Attention loops over batch reusing per-batch BD/scores/attn scratch.

#define BB 8
#define TT 1024
#define DD 512
#define HH 8
#define DKK 64

typedef __attribute__((ext_vector_type(16))) _Float16 v16h;
typedef __attribute__((ext_vector_type(8)))  float    v8f;

// ---------------- WMMA helpers ----------------

// Load a 16x32 (rows x K) fragment. `base` is row-major with leading dim `ld`
// (in halves). Each lane grabs 16 contiguous halves:
//   row = row0 + (lane & 15), k = col0 + (lane >> 4) * 16 .. +15
// Using the same contiguous scheme for A (MxK) and B (NxK transposed view)
// applies an identical K-permutation to both operands -> correct contraction.
__device__ __forceinline__ v16h load_frag(const _Float16* base, int row0, int col0,
                                          int ld, int lane) {
  const _Float16* p = base + (size_t)(row0 + (lane & 15)) * ld + col0 + ((lane >> 4) << 4);
  return *reinterpret_cast<const v16h*>(p);
}

__device__ __forceinline__ v8f wmma_f16(v16h a, v16h b, v8f c) {
  return __builtin_amdgcn_wmma_f32_16x16x32_f16(false, a, false, b, (short)0, c,
                                                false, false);
}

// C/D layout (16x16 f32, 8 VGPRs): m = r + (lane>>4)*8, n = lane & 15

// ---------------- CDNA5 async global->LDS copy (ASYNCcnt) ----------------
// Each lane copies 16 bytes: LDS[ldsoff] = MEM[gaddr]. VDST carries the
// per-lane LDS byte offset; VADDR is the 64-bit global address ("GV" mode).
__device__ __forceinline__ void async_copy_b128(unsigned ldsoff,
                                                const _Float16* gaddr) {
  asm volatile("global_load_async_to_lds_b128 %0, %1, off"
               :: "v"(ldsoff), "v"(gaddr) : "memory");
}
__device__ __forceinline__ void wait_async0() {
  asm volatile("s_wait_asynccnt 0x0" ::: "memory");
}

// ---------------- wave reductions (wave32) ----------------
__device__ __forceinline__ float wave_sum(float v) {
#pragma unroll
  for (int off = 16; off > 0; off >>= 1) v += __shfl_xor(v, off, 32);
  return v;
}
__device__ __forceinline__ float wave_max(float v) {
#pragma unroll
  for (int off = 16; off > 0; off >>= 1) v = fmaxf(v, __shfl_xor(v, off, 32));
  return v;
}

// ---------------- prep: weight f32->f16 + positional embedding ----------------
__global__ __launch_bounds__(256) void prep_kernel(
    const float* __restrict__ inputs, const float* __restrict__ wqkv,
    const float* __restrict__ wpos, const float* __restrict__ wout,
    _Float16* __restrict__ wqkv_h, _Float16* __restrict__ wpos_h,
    _Float16* __restrict__ wout_h, _Float16* __restrict__ pose_h) {
  constexpr int NQ = 3 * DD * DD, NP = DD * DD, NO = DD * DD, NE = TT * DD;
  const int total = NQ + NP + NO + NE;
  for (int idx = blockIdx.x * blockDim.x + threadIdx.x; idx < total;
       idx += gridDim.x * blockDim.x) {
    if (idx < NQ) {
      wqkv_h[idx] = (_Float16)wqkv[idx];
    } else if (idx < NQ + NP) {
      int i = idx - NQ; wpos_h[i] = (_Float16)wpos[i];
    } else if (idx < NQ + NP + NO) {
      int i = idx - NQ - NP; wout_h[i] = (_Float16)wout[i];
    } else {
      int i = idx - NQ - NP - NO;
      int t = i >> 9, c = i & 511;
      float ps = inputs[(size_t)t * DD];  // inputs[0, t, 0] acts as "position"
      int j = (c < 256) ? c : (c - 256);
      // inv_freq[j] = 10000^(-(2j)/512)
      float inv = expf(-(float)(2 * j) * (1.0f / 512.0f) * 9.210340371976184f);
      float ang = ps * inv;
      pose_h[i] = (_Float16)((c < 256) ? sinf(ang) : cosf(ang));
    }
  }
}

// ---------------- double LayerNorm (one wave per row of 512) ----------------
__global__ __launch_bounds__(256) void ln_kernel(
    const float* __restrict__ x, const float* __restrict__ g1,
    const float* __restrict__ b1, const float* __restrict__ g2,
    const float* __restrict__ b2, _Float16* __restrict__ xn) {
  int wid = threadIdx.x >> 5, lane = threadIdx.x & 31;
  int row = blockIdx.x * 8 + wid;  // 8192 rows
  const float* xr = x + (size_t)row * DD;
  float v[16], s = 0.f, ss = 0.f;
#pragma unroll
  for (int e = 0; e < 16; e++) {
    float t = xr[lane + 32 * e];
    v[e] = t; s += t; ss += t * t;
  }
  s = wave_sum(s); ss = wave_sum(ss);
  float mu = s * (1.0f / DD), var = ss * (1.0f / DD) - mu * mu;
  float rs = rsqrtf(var + 1e-5f);
  float s2 = 0.f, ss2 = 0.f;
#pragma unroll
  for (int e = 0; e < 16; e++) {
    int c = lane + 32 * e;
    float t = (v[e] - mu) * rs * g1[c] + b1[c];
    v[e] = t; s2 += t; ss2 += t * t;
  }
  s2 = wave_sum(s2); ss2 = wave_sum(ss2);
  float mu2 = s2 * (1.0f / DD), var2 = ss2 * (1.0f / DD) - mu2 * mu2;
  float rs2 = rsqrtf(var2 + 1e-5f);
  _Float16* o = xn + (size_t)row * DD;
#pragma unroll
  for (int e = 0; e < 16; e++) {
    int c = lane + 32 * e;
    o[c] = (_Float16)((v[e] - mu2) * rs2 * g2[c] + b2[c]);
  }
}

// ------------- QKV GEMM: (8192x512) @ (1536x512)^T ---------------------------
// Block = 256 threads (8 waves) computing a 128x64 tile. The 64x32-half weight
// panel is double-buffered in LDS via async copies; each wave owns 16 rows.
__global__ __launch_bounds__(256) void qkv_gemm_kernel(
    const _Float16* __restrict__ xn, const _Float16* __restrict__ w,
    const float* __restrict__ ubias, const float* __restrict__ vbias,
    _Float16* __restrict__ qu, _Float16* __restrict__ qv,
    _Float16* __restrict__ kk, _Float16* __restrict__ vT) {
  __shared__ __align__(32) _Float16 smB[2][64 * 32];
  int tid = threadIdx.x, wid = tid >> 5, lane = tid & 31;
  int bm = blockIdx.x / 24, bn = blockIdx.x % 24;  // 64 x 24 blocks
  int row0 = bm * 128 + wid * 16, col0 = bn * 64;

  // async-copy mapping: thread t moves 8 halves (16B); panel row = t/4
  int prow = tid >> 2, pk = (tid & 3) * 8;
  unsigned lo0 = (unsigned)(size_t)(&smB[0][tid * 8]);
  unsigned lo1 = (unsigned)(size_t)(&smB[1][tid * 8]);
  const _Float16* gB = w + (size_t)(col0 + prow) * DD + pk;

  async_copy_b128(lo0, gB);
  v8f c[4] = {};
  for (int step = 0; step < 16; ++step) {
    int cur = step & 1;
    wait_async0();
    __syncthreads();
    if (step + 1 < 16) async_copy_b128(cur ? lo0 : lo1, gB + (step + 1) * 32);
    v16h a = load_frag(xn, row0, step * 32, DD, lane);
    const _Float16* Bp = &smB[cur][0];
#pragma unroll
    for (int s = 0; s < 4; s++) {
      v16h b = *reinterpret_cast<const v16h*>(
          Bp + (s * 16 + (lane & 15)) * 32 + ((lane >> 4) << 4));
      c[s] = wmma_f16(a, b, c[s]);
    }
  }

  int mbase = (lane >> 4) << 3, n = lane & 15;
#pragma unroll
  for (int s = 0; s < 4; s++) {
    int col = col0 + s * 16 + n;
#pragma unroll
    for (int r = 0; r < 8; r++) {
      int rowg = row0 + mbase + r;
      int t = rowg & (TT - 1), bb = rowg >> 10;
      float val = c[s][r];
      if (col < DD) {                       // Q -> qu/qv with biases
        int h = col >> 6, dk = col & 63;
        size_t o = (((size_t)bb * HH + h) * TT + t) * DKK + dk;
        qu[o] = (_Float16)(val + ubias[col]);
        qv[o] = (_Float16)(val + vbias[col]);
      } else if (col < 2 * DD) {            // K
        int cc = col - DD; int h = cc >> 6, dk = cc & 63;
        kk[(((size_t)bb * HH + h) * TT + t) * DKK + dk] = (_Float16)val;
      } else {                              // V, stored transposed [h][dk][t]
        int cc = col - 2 * DD; int h = cc >> 6, dk = cc & 63;
        vT[(((size_t)bb * HH + h) * DKK + dk) * TT + t] = (_Float16)val;
      }
    }
  }
}

// ------------- pos GEMM: pos_emb(1024x512) @ W_pos^T -> p (T, H*DK) ----------
__global__ __launch_bounds__(256) void pos_gemm_kernel(
    const _Float16* __restrict__ pe, const _Float16* __restrict__ w,
    _Float16* __restrict__ p) {
  int wid = threadIdx.x >> 5, lane = threadIdx.x & 31;
  int wg = blockIdx.x * 8 + wid;  // 512 waves
  int mt = wg >> 3, nt = wg & 7;
  int row0 = mt * 16, col0 = nt * 64;
  v8f c[4] = {};
  for (int k0 = 0; k0 < DD; k0 += 32) {
    v16h a = load_frag(pe, row0, k0, DD, lane);
#pragma unroll
    for (int s = 0; s < 4; s++) {
      v16h b = load_frag(w, col0 + s * 16, k0, DD, lane);
      c[s] = wmma_f16(a, b, c[s]);
    }
  }
  int mbase = (lane >> 4) << 3, n = lane & 15;
#pragma unroll
  for (int s = 0; s < 4; s++)
#pragma unroll
    for (int r = 0; r < 8; r++)
      p[(size_t)(row0 + mbase + r) * DD + col0 + s * 16 + n] = (_Float16)c[s][r];
}

// ------------- BD = qv . p^T for one batch: [H,T,T] f16 ----------------------
__global__ __launch_bounds__(256) void bd_gemm_kernel(
    const _Float16* __restrict__ qv, const _Float16* __restrict__ p,
    _Float16* __restrict__ bd) {
  int wid = threadIdx.x >> 5, lane = threadIdx.x & 31;
  int wg = blockIdx.x * 8 + wid;          // 8192 waves
  int h = wg >> 10, rem = wg & 1023;
  int rt = rem >> 4, mt = rem & 15;
  int row0 = rt * 16, col0 = mt * 64;
  const _Float16* A = qv + (size_t)h * TT * DKK;
  v8f c[4] = {};
  for (int k0 = 0; k0 < DKK; k0 += 32) {
    v16h a = load_frag(A, row0, k0, DKK, lane);
#pragma unroll
    for (int s = 0; s < 4; s++) {
      v16h b = load_frag(p, col0 + s * 16, h * DKK + k0, DD, lane);
      c[s] = wmma_f16(a, b, c[s]);
    }
  }
  _Float16* O = bd + (size_t)h * TT * TT;
  int mbase = (lane >> 4) << 3, n = lane & 15;
#pragma unroll
  for (int s = 0; s < 4; s++)
#pragma unroll
    for (int r = 0; r < 8; r++)
      O[(size_t)(row0 + mbase + r) * TT + col0 + s * 16 + n] = (_Float16)c[s][r];
}

// ------------- scores = (qu.k^T + rel_shift(BD)) / sqrt(DK), per batch -------
// rel_shift(BD)[i,j] = BD[i, T-1+j-i]       if j <= i
//                    = 0                     if j == i+1
//                    = BD[i+1, j-i-2]        if j >  i+1
__global__ __launch_bounds__(256) void scores_kernel(
    const _Float16* __restrict__ qu, const _Float16* __restrict__ kk,
    const _Float16* __restrict__ bd, float* __restrict__ sc) {
  int wid = threadIdx.x >> 5, lane = threadIdx.x & 31;
  int wg = blockIdx.x * 8 + wid;          // 8192 waves
  int h = wg >> 10, rem = wg & 1023;
  int it = rem >> 4, jt = rem & 15;
  int row0 = it * 16, col0 = jt * 64;
  const _Float16* A = qu + (size_t)h * TT * DKK;
  const _Float16* Bm = kk + (size_t)h * TT * DKK;
  v8f c[4] = {};
  for (int k0 = 0; k0 < DKK; k0 += 32) {
    v16h a = load_frag(A, row0, k0, DKK, lane);
#pragma unroll
    for (int s = 0; s < 4; s++) {
      v16h b = load_frag(Bm, col0 + s * 16, k0, DKK, lane);
      c[s] = wmma_f16(a, b, c[s]);
    }
  }
  const _Float16* BD = bd + (size_t)h * TT * TT;
  float* O = sc + (size_t)h * TT * TT;
  int mbase = (lane >> 4) << 3, n = lane & 15;
#pragma unroll
  for (int s = 0; s < 4; s++)
#pragma unroll
    for (int r = 0; r < 8; r++) {
      int i = row0 + mbase + r, j = col0 + s * 16 + n;
      float bv;
      if (j <= i)            bv = (float)BD[(size_t)i * TT + (TT - 1 + j - i)];
      else if (j == i + 1)   bv = 0.0f;
      else                   bv = (float)BD[(size_t)(i + 1) * TT + (j - i - 2)];
      O[(size_t)i * TT + j] = (c[s][r] + bv) * 0.125f;  // 1/sqrt(64)
    }
}

// ------------- softmax over rows of 1024 (one wave per row) ------------------
__global__ __launch_bounds__(256) void softmax_kernel(
    const float* __restrict__ sc, _Float16* __restrict__ at) {
  int wid = threadIdx.x >> 5, lane = threadIdx.x & 31;
  int row = blockIdx.x * 8 + wid;  // H*T rows
  const float* r = sc + (size_t)row * TT;
  float v[32], mx = -3.4e38f;
#pragma unroll
  for (int e = 0; e < 32; e++) {
    float t = r[lane + 32 * e];
    v[e] = t; mx = fmaxf(mx, t);
  }
  mx = wave_max(mx);
  float s = 0.f;
#pragma unroll
  for (int e = 0; e < 32; e++) {
    float t = __expf(v[e] - mx);
    v[e] = t; s += t;
  }
  s = wave_sum(s);
  float inv = 1.0f / s;
  _Float16* o = at + (size_t)row * TT;
#pragma unroll
  for (int e = 0; e < 32; e++) o[lane + 32 * e] = (_Float16)(v[e] * inv);
}

// ------------- O = attn . V (per batch), write [t, h*DK+d] as f16 ------------
__global__ __launch_bounds__(256) void pv_kernel(
    const _Float16* __restrict__ at, const _Float16* __restrict__ vT,
    _Float16* __restrict__ ao) {
  int wid = threadIdx.x >> 5, lane = threadIdx.x & 31;
  int wg = blockIdx.x * 8 + wid;   // 512 waves
  int h = wg >> 6, it = wg & 63;
  int row0 = it * 16;
  const _Float16* A = at + (size_t)h * TT * TT;
  const _Float16* Bm = vT + (size_t)h * DKK * TT;
  v8f c[4] = {};
  for (int k0 = 0; k0 < TT; k0 += 32) {
    v16h a = load_frag(A, row0, k0, TT, lane);
#pragma unroll
    for (int s = 0; s < 4; s++) {
      v16h b = load_frag(Bm, s * 16, k0, TT, lane);
      c[s] = wmma_f16(a, b, c[s]);
    }
  }
  int mbase = (lane >> 4) << 3, n = lane & 15;
#pragma unroll
  for (int s = 0; s < 4; s++)
#pragma unroll
    for (int r = 0; r < 8; r++)
      ao[(size_t)(row0 + mbase + r) * DD + h * DKK + s * 16 + n] =
          (_Float16)c[s][r];
}

// ------------- out = attn_out @ W_out^T + b_out (f32 result) -----------------
// Same async-LDS-staged block structure as the QKV GEMM (128x64 block tile).
__global__ __launch_bounds__(256) void outproj_kernel(
    const _Float16* __restrict__ ao, const _Float16* __restrict__ w,
    const float* __restrict__ bias, float* __restrict__ out) {
  __shared__ __align__(32) _Float16 smB[2][64 * 32];
  int tid = threadIdx.x, wid = tid >> 5, lane = tid & 31;
  int bm = blockIdx.x >> 3, bn = blockIdx.x & 7;  // 64 x 8 blocks
  int row0 = bm * 128 + wid * 16, col0 = bn * 64;

  int prow = tid >> 2, pk = (tid & 3) * 8;
  unsigned lo0 = (unsigned)(size_t)(&smB[0][tid * 8]);
  unsigned lo1 = (unsigned)(size_t)(&smB[1][tid * 8]);
  const _Float16* gB = w + (size_t)(col0 + prow) * DD + pk;

  async_copy_b128(lo0, gB);
  v8f c[4] = {};
  for (int step = 0; step < 16; ++step) {
    int cur = step & 1;
    wait_async0();
    __syncthreads();
    if (step + 1 < 16) async_copy_b128(cur ? lo0 : lo1, gB + (step + 1) * 32);
    v16h a = load_frag(ao, row0, step * 32, DD, lane);
    const _Float16* Bp = &smB[cur][0];
#pragma unroll
    for (int s = 0; s < 4; s++) {
      v16h b = *reinterpret_cast<const v16h*>(
          Bp + (s * 16 + (lane & 15)) * 32 + ((lane >> 4) << 4));
      c[s] = wmma_f16(a, b, c[s]);
    }
  }

  int mbase = (lane >> 4) << 3, n = lane & 15;
#pragma unroll
  for (int s = 0; s < 4; s++)
#pragma unroll
    for (int r = 0; r < 8; r++) {
      int col = col0 + s * 16 + n;
      out[(size_t)(row0 + mbase + r) * DD + col] = c[s][r] + bias[col];
    }
}

// ---------------- launcher ----------------
extern "C" void kernel_launch(void* const* d_in, const int* in_sizes, int n_in,
                              void* d_out, int out_size, void* d_ws, size_t ws_size,
                              hipStream_t stream) {
  (void)in_sizes; (void)n_in; (void)out_size; (void)ws_size;
  const float* inputs = (const float*)d_in[0];
  const float* ln1_g  = (const float*)d_in[1];
  const float* ln1_b  = (const float*)d_in[2];
  const float* ln2_g  = (const float*)d_in[3];
  const float* ln2_b  = (const float*)d_in[4];
  const float* W_qkv  = (const float*)d_in[5];
  const float* W_pos  = (const float*)d_in[6];
  const float* W_out  = (const float*)d_in[7];
  const float* b_out  = (const float*)d_in[8];
  const float* pbu    = (const float*)d_in[9];
  const float* pbv    = (const float*)d_in[10];
  float* out = (float*)d_out;

  char* ws = (char*)d_ws;
  size_t off = 0;
  auto alloc = [&](size_t bytes) -> char* {
    char* p = ws + off;
    off += (bytes + 255) & ~(size_t)255;
    return p;
  };
  _Float16* wqkv_h = (_Float16*)alloc((size_t)3 * DD * DD * 2);
  _Float16* wpos_h = (_Float16*)alloc((size_t)DD * DD * 2);
  _Float16* wout_h = (_Float16*)alloc((size_t)DD * DD * 2);
  _Float16* pose_h = (_Float16*)alloc((size_t)TT * DD * 2);
  _Float16* xn_h   = (_Float16*)alloc((size_t)BB * TT * DD * 2);
  _Float16* qu_h   = (_Float16*)alloc((size_t)BB * HH * TT * DKK * 2);
  _Float16* qv_h   = (_Float16*)alloc((size_t)BB * HH * TT * DKK * 2);
  _Float16* k_h    = (_Float16*)alloc((size_t)BB * HH * TT * DKK * 2);
  _Float16* vT_h   = (_Float16*)alloc((size_t)BB * HH * DKK * TT * 2);
  _Float16* p_h    = (_Float16*)alloc((size_t)TT * DD * 2);
  _Float16* ao_h   = (_Float16*)alloc((size_t)BB * TT * DD * 2);
  _Float16* bd_h   = (_Float16*)alloc((size_t)HH * TT * TT * 2);   // per-batch
  float*    sc_f   = (float*)   alloc((size_t)HH * TT * TT * 4);   // per-batch
  _Float16* at_h   = (_Float16*)alloc((size_t)HH * TT * TT * 2);   // per-batch

  prep_kernel<<<1792, 256, 0, stream>>>(inputs, W_qkv, W_pos, W_out,
                                        wqkv_h, wpos_h, wout_h, pose_h);
  ln_kernel<<<1024, 256, 0, stream>>>(inputs, ln1_g, ln1_b, ln2_g, ln2_b, xn_h);
  qkv_gemm_kernel<<<1536, 256, 0, stream>>>(xn_h, wqkv_h, pbu, pbv,
                                            qu_h, qv_h, k_h, vT_h);
  pos_gemm_kernel<<<64, 256, 0, stream>>>(pose_h, wpos_h, p_h);

  for (int b = 0; b < BB; b++) {
    const _Float16* qu_b = qu_h + (size_t)b * HH * TT * DKK;
    const _Float16* qv_b = qv_h + (size_t)b * HH * TT * DKK;
    const _Float16* k_b  = k_h  + (size_t)b * HH * TT * DKK;
    const _Float16* vT_b = vT_h + (size_t)b * HH * DKK * TT;
    _Float16* ao_b = ao_h + (size_t)b * TT * DD;
    bd_gemm_kernel<<<1024, 256, 0, stream>>>(qv_b, p_h, bd_h);
    scores_kernel<<<1024, 256, 0, stream>>>(qu_b, k_b, bd_h, sc_f);
    softmax_kernel<<<1024, 256, 0, stream>>>(sc_f, at_h);
    pv_kernel<<<64, 256, 0, stream>>>(at_h, vT_b, ao_b);
  }

  outproj_kernel<<<512, 256, 0, stream>>>(ao_h, wout_h, b_out, out);
}
